// MatchingLayerL2_71116068487480
// MI455X (gfx1250) — matrix operand order repaired and checked by reading, not compiled, not run on hardware
//
#include <hip/hip_runtime.h>

typedef __attribute__((ext_vector_type(16))) _Float16 v16h;
typedef __attribute__((ext_vector_type(8)))  _Float16 v8h;
typedef __attribute__((ext_vector_type(4)))  _Float16 v4h;
typedef __attribute__((ext_vector_type(8)))  float    v8f;

#define DDIM 128
#define EPSF 1e-12f
#define LOG2E 1.44269504088896340736f

// ---------------------------------------------------------------------------
// Kernel 1: fp32 rows -> f16 rows (row-major) + per-row squared L2 norm.
// One wave (32 lanes) per row, 4 elements per lane.
// ---------------------------------------------------------------------------
__global__ __launch_bounds__(256) void cvt_norm_kernel(
    const float* __restrict__ src, _Float16* __restrict__ dst,
    float* __restrict__ nrm, int rows)
{
  const int row  = blockIdx.x * 8 + (threadIdx.x >> 5);
  const int lane = threadIdx.x & 31;
  if (row >= rows) return;

  const float4 f = ((const float4*)(src + (size_t)row * DDIM))[lane];
  v4h h;
  h[0] = (_Float16)f.x; h[1] = (_Float16)f.y;
  h[2] = (_Float16)f.z; h[3] = (_Float16)f.w;
  ((v4h*)(dst + (size_t)row * DDIM))[lane] = h;

  float ss = fmaf(f.x, f.x, fmaf(f.y, f.y, fmaf(f.z, f.z, f.w * f.w)));
  #pragma unroll
  for (int m = 1; m < 32; m <<= 1) ss += __shfl_xor(ss, m, 32);
  if (lane == 0) nrm[row] = ss;
}

// ---------------------------------------------------------------------------
// Kernel 2: istd[j] = 1 / std[j]  (precise divide off the hot path).
// ---------------------------------------------------------------------------
__global__ __launch_bounds__(256) void rcp_kernel(
    const float* __restrict__ s, float* __restrict__ r, int n)
{
  const int i = blockIdx.x * 256 + threadIdx.x;
  if (i < n) r[i] = 1.0f / s[i];
}

// ---------------------------------------------------------------------------
// B-fragment load for one 16x16 tile: 8x global_load_b128 into one buffer.
// B layout: lane l = column l&15; h[0..7] = K=kb*32+hi*8.., h[8..15] = +16.
// ---------------------------------------------------------------------------
__device__ __forceinline__ void load_B(const _Float16* __restrict__ yrow,
                                       int hi, v16h (&B)[4])
{
  #pragma unroll
  for (int kb = 0; kb < 4; ++kb) {
    const int k0 = kb * 32 + hi * 8;
    v8h lo = *(const v8h*)(yrow + k0);
    v8h hh = *(const v8h*)(yrow + k0 + 16);
    B[kb] = __builtin_shufflevector(lo, hh,
            0,1,2,3,4,5,6,7,8,9,10,11,12,13,14,15);
  }
}

__device__ __forceinline__ v8f wmma_chain(const v16h (&A)[4], const v16h (&B)[4])
{
  v8f acc = {};
  #pragma unroll
  for (int kb = 0; kb < 4; ++kb)
    acc = __builtin_amdgcn_wmma_f32_16x16x32_f16(
            false, A[kb], false, B[kb], (short)0, acc, false, false);
  return acc;
}

// Pass-A epilogue: Z += exp(sim) computed as exp2(-sqrt(d2) * istd*log2e).
__device__ __forceinline__ void tile_math_A(const v8f& acc, const float (&a_r)[8],
                                            float bj, float isl, float (&s)[8])
{
  #pragma unroll
  for (int r = 0; r < 8; ++r) {
    const float d2 = fmaf(-2.0f, acc[r], a_r[r] + bj);
    const float sv = __builtin_amdgcn_sqrtf(fmaxf(d2, EPSF));
    s[r] += __builtin_amdgcn_exp2f(-sv * isl);
  }
}

// Pass-B epilogue: out = -sqrt(d2)*istd - logZ = fma(-sqrt, istd, -logZ).
__device__ __forceinline__ void tile_math_B(const v8f& acc, const float (&a_r)[8],
                                            float bj, float is, const float (&nlz)[8],
                                            float* __restrict__ op, int M)
{
  #pragma unroll
  for (int r = 0; r < 8; ++r) {
    const float d2 = fmaf(-2.0f, acc[r], a_r[r] + bj);
    const float sv = __builtin_amdgcn_sqrtf(fmaxf(d2, EPSF));
    op[(size_t)r * M] = fmaf(-sv, is, nlz[r]);
  }
}

// ---------------------------------------------------------------------------
// Kernel 3: fused  dist -> sim -> log_softmax  with recompute (2 GEMM passes).
// Block = 16 waves, owns 16 rows. Wave w owns columns [w*M/16, (w+1)*M/16).
// B fragments are double-buffered (2 tiles in flight) to hide L2 latency
// behind the WMMA chain + epilogue of the previous tile.
// ---------------------------------------------------------------------------
__global__ __launch_bounds__(512) void l2_logsoftmax_kernel(
    const _Float16* __restrict__ xh, const _Float16* __restrict__ yh,
    const float* __restrict__ anorm, const float* __restrict__ bnorm,
    const float* __restrict__ istd, float* __restrict__ out, int M)
{
  __shared__ float s_part[16][16];   // [wave][row] partial Z
  __shared__ float s_logZ[16];

  const int tid  = threadIdx.x;
  const int wave = tid >> 5;
  const int lane = tid & 31;
  const int l16  = lane & 15;
  const int hi   = lane >> 4;              // 0 or 1 (half-wave)
  const int iBase = blockIdx.x * 16;

  // ---- A tile (16 rows x K=128), ISA 16-bit A layout:
  // lane l (row = l&15): per 32-K block, h[0..7] = K=kb*32+hi*8.., h[8..15] = +16.
  v16h A[4];
  {
    const _Float16* xrow = xh + (size_t)(iBase + l16) * DDIM;
    #pragma unroll
    for (int kb = 0; kb < 4; ++kb) {
      const int k0 = kb * 32 + hi * 8;
      v8h lo = *(const v8h*)(xrow + k0);
      v8h hh = *(const v8h*)(xrow + k0 + 16);
      A[kb] = __builtin_shufflevector(lo, hh,
              0,1,2,3,4,5,6,7,8,9,10,11,12,13,14,15);
    }
  }

  // ||x_i||^2 for the 8 rows this lane's accumulator covers (broadcast loads).
  float a_r[8];
  #pragma unroll
  for (int r = 0; r < 8; ++r) a_r[r] = anorm[iBase + hi * 8 + r];

  const int jChunk = M >> 4;               // columns per wave (2048)
  const int j0 = wave * jChunk;
  const int nTiles = jChunk >> 4;          // 16x16 tiles per wave (128, even)

  float s[8];
  #pragma unroll
  for (int r = 0; r < 8; ++r) s[r] = 0.0f;

  // ================= pass A: Z_row = sum_j exp(sim) =================
  // sim <= 0 always (dist >= 0, std > 0) so no max-shift is needed in fp32.
  {
    const _Float16* yrow = yh + (size_t)(j0 + l16) * DDIM;
    const float* bp = bnorm + (j0 + l16);
    const float* ip = istd  + (j0 + l16);
    v16h B0[4], B1[4];
    load_B(yrow, hi, B0);
    for (int t = 0; t < nTiles; t += 2) {
      load_B(yrow + 16 * DDIM, hi, B1);            // tile t+1 in flight
      __builtin_prefetch(yrow + 64 * DDIM, 0, 1);
      v8f acc = wmma_chain(A, B0);
      tile_math_A(acc, a_r, bp[0], ip[0] * LOG2E, s);
      load_B(yrow + 32 * DDIM, hi, B0);            // tile t+2 in flight
      acc = wmma_chain(A, B1);
      tile_math_A(acc, a_r, bp[16], ip[16] * LOG2E, s);
      yrow += 32 * DDIM; bp += 32; ip += 32;
    }
  }

  // Reduce the 16 column-lanes of each half-wave (rows hi*8..hi*8+7).
  #pragma unroll
  for (int r = 0; r < 8; ++r) {
    #pragma unroll
    for (int m = 1; m < 16; m <<= 1) s[r] += __shfl_xor(s[r], m, 32);
  }
  if (l16 == 0) {
    #pragma unroll
    for (int r = 0; r < 8; ++r) s_part[wave][hi * 8 + r] = s[r];
  }
  __syncthreads();
  if (tid < 16) {                           // waves cover disjoint columns: add
    float z = 0.0f;
    #pragma unroll
    for (int w = 0; w < 16; ++w) z += s_part[w][tid];
    s_logZ[tid] = __logf(z);
  }
  __syncthreads();

  float nlz[8];                             // -logZ for this lane's 8 rows
  #pragma unroll
  for (int r = 0; r < 8; ++r) nlz[r] = -s_logZ[hi * 8 + r];

  // ================= pass B: out = sim - logZ =================
  {
    const _Float16* yrow = yh + (size_t)(j0 + l16) * DDIM;
    const float* bp = bnorm + (j0 + l16);
    const float* ip = istd  + (j0 + l16);
    float* op = out + (size_t)(iBase + hi * 8) * M + (j0 + l16);
    v16h B0[4], B1[4];
    load_B(yrow, hi, B0);
    for (int t = 0; t < nTiles; t += 2) {
      load_B(yrow + 16 * DDIM, hi, B1);
      __builtin_prefetch(yrow + 64 * DDIM, 0, 1);
      v8f acc = wmma_chain(A, B0);
      tile_math_B(acc, a_r, bp[0], ip[0], nlz, op, M);
      load_B(yrow + 32 * DDIM, hi, B0);
      acc = wmma_chain(A, B1);
      tile_math_B(acc, a_r, bp[16], ip[16], nlz, op + 16, M);
      yrow += 32 * DDIM; bp += 32; ip += 32; op += 32;
    }
  }
}

// ---------------------------------------------------------------------------
// Workspace layout (bytes):
//   yh    : M*128*2  = 8 MiB     (f16 copy of y)
//   xh    : N*128*2  = 1 MiB     (f16 copy of x)   [also absorbs pipeline overread]
//   bnorm : M*4      = 128 KiB
//   anorm : N*4      = 16 KiB
//   istd  : M*4      = 128 KiB
// total ~ 9.5 MiB.
// ---------------------------------------------------------------------------
extern "C" void kernel_launch(void* const* d_in, const int* in_sizes, int n_in,
                              void* d_out, int out_size, void* d_ws, size_t ws_size,
                              hipStream_t stream) {
  const float* x    = (const float*)d_in[0];
  const float* y    = (const float*)d_in[1];
  const float* stdv = (const float*)d_in[2];
  const int N = in_sizes[0] / DDIM;   // 4096
  const int M = in_sizes[2];          // 32768

  char* ws = (char*)d_ws;
  _Float16* yh = (_Float16*)ws;
  _Float16* xh = (_Float16*)(ws + (size_t)M * DDIM * 2);
  float* bnorm = (float*)(ws + (size_t)M * DDIM * 2 + (size_t)N * DDIM * 2);
  float* anorm = bnorm + M;
  float* istd  = anorm + N;

  cvt_norm_kernel<<<(M + 7) / 8, 256, 0, stream>>>(y, yh, bnorm, M);
  cvt_norm_kernel<<<(N + 7) / 8, 256, 0, stream>>>(x, xh, anorm, N);
  rcp_kernel<<<(M + 255) / 256, 256, 0, stream>>>(stdv, istd, M);

  l2_logsoftmax_kernel<<<N / 16, 512, 0, stream>>>(
      xh, yh, anorm, bnorm, istd, (float*)d_out, M);

  (void)n_in; (void)out_size; (void)ws_size;
}
